// MotionFormer_5291399708693
// MI455X (gfx1250) — matrix operand order repaired and checked by reading, not compiled, not run on hardware
//
#include <hip/hip_runtime.h>
#include <hip/hip_bf16.h>

typedef __attribute__((ext_vector_type(16))) _Float16 v16h;
typedef __attribute__((ext_vector_type(8)))  _Float16 v8h;
typedef __attribute__((ext_vector_type(8)))  float    v8f;
typedef _Float16 half_t;

#define R_TOT 16384   // B*N
#define CDIM 256
#define NPTS 1024
#define HID 1024
#define MOT 64
#define ATT_SCALE 0.17677669529663687f  // 1/sqrt(32), folded into q projection

// ---------------- WMMA fragment helpers (wave32; contiguous k = 16*(lane/16)+s for both A and B) ----------

__device__ inline v8f wmma_f16(v16h a, v16h b, v8f c) {
  return __builtin_amdgcn_wmma_f32_16x16x32_f16(false, a, false, b, (short)0, c, false, false);
}

// A fragment: 16x32 f16 from row-major source (lda in halves, 16B-aligned rows).
// Per lane: rows m=lane&15; halves k = hi*16..hi*16+15 contiguous -> two b128 loads.
__device__ inline v16h load_a_frag(const half_t* a, int lda) {
  int lane = threadIdx.x & 31;
  int m = lane & 15;
  int hi = (lane >> 4) & 1;
  const half_t* row = a + (size_t)m * lda + hi * 16;
  v8h x0 = *(const v8h*)(row);
  v8h x1 = *(const v8h*)(row + 8);
  return __builtin_shufflevector(x0, x1, 0, 1, 2, 3, 4, 5, 6, 7,
                                 8, 9, 10, 11, 12, 13, 14, 15);
}

// B fragment 32x16 from an N-major (B^T) source: bt[n][k], ldk in halves.
// Per lane: col n=lane&15, halves k = kh*16..kh*16+15 contiguous -> two b128 loads.
__device__ inline v16h load_bt_frag(const half_t* bt, int ldk) {
  int lane = threadIdx.x & 31;
  int n = lane & 15;
  int kh = (lane >> 4) & 1;
  const half_t* p = bt + (size_t)n * ldk + kh * 16;
  v8h x0 = *(const v8h*)p;
  v8h x1 = *(const v8h*)(p + 8);
  return __builtin_shufflevector(x0, x1, 0, 1, 2, 3, 4, 5, 6, 7,
                                 8, 9, 10, 11, 12, 13, 14, 15);
}

// B^T fragment with column padding (cols >= nvalid read as 0)
__device__ inline v16h load_bt_frag_pad(const half_t* bt, int ldk, int nvalid) {
  int lane = threadIdx.x & 31;
  int n = lane & 15;
  int kh = (lane >> 4) & 1;
  v16h f = {};
  if (n < nvalid) {
    const half_t* p = bt + (size_t)n * ldk + kh * 16;
    v8h x0 = *(const v8h*)p;
    v8h x1 = *(const v8h*)(p + 8);
    f = __builtin_shufflevector(x0, x1, 0, 1, 2, 3, 4, 5, 6, 7,
                                8, 9, 10, 11, 12, 13, 14, 15);
  }
  return f;
}

// ---------------- small elementwise / reduction kernels ----------------

// f32 [K][N] weight -> f16 transposed [N][K]
__global__ void cvt_transpose_h(const float* __restrict__ s, half_t* __restrict__ d,
                                int K, int N) {
  int i = blockIdx.x * blockDim.x + threadIdx.x;
  if (i < K * N) {
    int k = i / N, n = i - k * N;
    d[(size_t)n * K + k] = (half_t)s[i];
  }
}

// BN stats over (B,N) for x in [B,C,N] layout; one block per channel.
__global__ void bn_stats_bcn(const float* __restrict__ x, float* mu, float* rs) {
  int c = blockIdx.x, t = threadIdx.x;
  float s = 0.f, q = 0.f;
  for (int i = t; i < R_TOT; i += 256) {
    int b = i >> 10, n = i & 1023;
    float v = x[((size_t)(b * CDIM + c)) * NPTS + n];
    s += v; q += v * v;
  }
  __shared__ float ss[256], qq[256];
  ss[t] = s; qq[t] = q; __syncthreads();
  for (int o = 128; o; o >>= 1) {
    if (t < o) { ss[t] += ss[t + o]; qq[t] += qq[t + o]; }
    __syncthreads();
  }
  if (t == 0) {
    float m = ss[0] * (1.f / R_TOT);
    float var = qq[0] * (1.f / R_TOT) - m * m;
    mu[c] = m; rs[c] = rsqrtf(var + 1e-5f);
  }
}

// BN stats for xn2 in [R, C] layout
__global__ void bn_stats_rc(const float* __restrict__ x, float* mu, float* rs) {
  int c = blockIdx.x, t = threadIdx.x;
  float s = 0.f, q = 0.f;
  for (int i = t; i < R_TOT; i += 256) {
    float v = x[(size_t)i * CDIM + c];
    s += v; q += v * v;
  }
  __shared__ float ss[256], qq[256];
  ss[t] = s; qq[t] = q; __syncthreads();
  for (int o = 128; o; o >>= 1) {
    if (t < o) { ss[t] += ss[t + o]; qq[t] += qq[t + o]; }
    __syncthreads();
  }
  if (t == 0) {
    float m = ss[0] * (1.f / R_TOT);
    float var = qq[0] * (1.f / R_TOT) - m * m;
    mu[c] = m; rs[c] = rsqrtf(var + 1e-5f);
  }
}

// normalize+transpose x -> xn[R,C] (f32 + f16); ce = cor @ Wcor (f32 natural +
// f16 transposed per head: ceT[b][h][j][key])
__global__ void stage_norm1(const float* __restrict__ x, const float* __restrict__ cor,
                            const float* __restrict__ g1, const float* __restrict__ b1,
                            const float* __restrict__ mu, const float* __restrict__ rs,
                            const float* __restrict__ Wcor,
                            float* __restrict__ xn, half_t* __restrict__ xn_h,
                            float* __restrict__ ce, half_t* __restrict__ ceT) {
  int row = blockIdx.x;             // b*1024 + n
  int b = row >> 10, n = row & 1023;
  int c = threadIdx.x;
  float v = x[((size_t)(b * CDIM + c)) * NPTS + n];
  float y = (v - mu[c]) * rs[c] * g1[c] + b1[c];
  xn[(size_t)row * CDIM + c] = y;
  xn_h[(size_t)row * CDIM + c] = (half_t)y;
  if (c < MOT) {
    float c0 = cor[(size_t)row * 3 + 0];
    float c1 = cor[(size_t)row * 3 + 1];
    float c2 = cor[(size_t)row * 3 + 2];
    float e = c0 * Wcor[c] + c1 * Wcor[MOT + c] + c2 * Wcor[2 * MOT + c];
    ce[(size_t)row * MOT + c] = e;
    ceT[((size_t)(b * 8 + (c >> 3)) * 8 + (c & 7)) * NPTS + n] = (half_t)e;
  }
}

__global__ void bn2_apply(const float* __restrict__ xn2, const float* __restrict__ mu,
                          const float* __restrict__ rs, const float* __restrict__ g,
                          const float* __restrict__ bb, half_t* __restrict__ back_h) {
  size_t i = (size_t)blockIdx.x * blockDim.x + threadIdx.x;  // over R*C
  int c = (int)(i & (CDIM - 1));
  back_h[i] = (half_t)((xn2[i] - mu[c]) * rs[c] * g[c] + bb[c]);
}

// ---------------- generic WMMA GEMM: C[M,N] = A[M,K](f16) @ B[K,N](f16, stored B^T) ----
// block = 4 waves; wave w -> M tile (blockIdx.x*4+w)*16, N strip blockIdx.y*64.

constexpr int EPI_QSCALE = 0;        // outh = acc * ATT_SCALE (q projection)
constexpr int EPI_F32_BIAS = 1;      // outf = acc + bias[n]
constexpr int EPI_F32_BIAS_RES = 2;  // outf = acc + bias[n] + e0[m,n]
constexpr int EPI_FC1 = 3;           // outh = prelu((acc+bias)*e0 + e1), alpha=e2[0]
constexpr int EPI_FC2_TRANS = 4;     // outf[(b*C+n)*Np+p] = e0[same] + acc + bias[n]
constexpr int EPI_KV = 5;            // n<256 -> k_h natural; n>=256 -> vT[b][h][d][key]

template <int EPI>
__global__ void __launch_bounds__(128)
gemm_wmma(const half_t* __restrict__ A, const half_t* __restrict__ Bt,
          int M, int N, int K, float* __restrict__ outf, half_t* __restrict__ outh,
          half_t* __restrict__ outh2, const float* __restrict__ bias,
          const float* __restrict__ e0, const float* __restrict__ e1,
          const float* __restrict__ e2) {
  int wid = threadIdx.x >> 5;
  int m0 = (blockIdx.x * 4 + wid) * 16;
  int n0 = blockIdx.y * 64;
  v8f acc[4] = {};
  for (int k0 = 0; k0 < K; k0 += 32) {
    if (k0 + 32 < K)
      __builtin_prefetch(A + (size_t)(m0 + (threadIdx.x & 15)) * K + k0 + 32, 0, 3);
    v16h a = load_a_frag(A + (size_t)m0 * K + k0, K);
#pragma unroll
    for (int t = 0; t < 4; ++t) {
      v16h bf = load_bt_frag(Bt + (size_t)(n0 + t * 16) * K + k0, K);
      acc[t] = wmma_f16(a, bf, acc[t]);
    }
  }
  int lane = threadIdx.x & 31;
  int n = lane & 15, mb = (lane >> 4) * 8;
#pragma unroll
  for (int t = 0; t < 4; ++t) {
    int nn = n0 + t * 16 + n;
#pragma unroll
    for (int r = 0; r < 8; ++r) {
      int mm = m0 + mb + r;
      float v = acc[t][r];
      if constexpr (EPI == EPI_QSCALE) {
        outh[(size_t)mm * N + nn] = (half_t)(v * ATT_SCALE);
      } else if constexpr (EPI == EPI_F32_BIAS) {
        outf[(size_t)mm * N + nn] = v + bias[nn];
      } else if constexpr (EPI == EPI_F32_BIAS_RES) {
        outf[(size_t)mm * N + nn] = v + bias[nn] + e0[(size_t)mm * N + nn];
      } else if constexpr (EPI == EPI_FC1) {
        float t2 = (v + bias[nn]) * e0[nn] + e1[nn];
        t2 = t2 >= 0.f ? t2 : e2[0] * t2;
        outh[(size_t)mm * N + nn] = (half_t)t2;
      } else if constexpr (EPI == EPI_FC2_TRANS) {
        int b = mm >> 10, pnt = mm & 1023;
        size_t oi = ((size_t)(b * CDIM + nn)) * NPTS + pnt;
        outf[oi] = e0[oi] + v + bias[nn];
      } else if constexpr (EPI == EPI_KV) {
        half_t hv = (half_t)v;
        if (nn < 256) {
          outh[(size_t)mm * 256 + nn] = hv;  // k natural [key][dim]
        } else {
          int d = nn - 256;
          int b = mm >> 10, key = mm & 1023;
          outh2[((size_t)(b * 8 + (d >> 5)) * 32 + (d & 31)) * NPTS + key] = hv;
        }
      }
    }
  }
}

// ---------------- fused inter-frame attention ----------------
// one wave per (qtile, head, batch). QK^T -> LDS f32 (scale pre-folded into q),
// softmax (unnormalized exp stored f16, 1/sum folded into epilogue),
// PV + P*ce_head via WMMA.

__global__ void __launch_bounds__(32)
attn_fused(const half_t* __restrict__ q_h, const half_t* __restrict__ k_h,
           const half_t* __restrict__ vT, const half_t* __restrict__ ceT,
           const float* __restrict__ ce,
           half_t* __restrict__ xattn_h, half_t* __restrict__ mdiff_h) {
  __shared__ float  s_lds[16 * 1024];
  __shared__ half_t p_lds[16 * 1024];
  __shared__ float  inv_lds[16];

  int qt = blockIdx.x, h = blockIdx.y, b = blockIdx.z;
  int b2 = (b + 8) & 15;          // cross-frame swapped batch
  int q0 = qt * 16;
  int lane = threadIdx.x;
  int nidx = lane & 15, mbase = (lane >> 4) * 8;

  // --- scores: S = (q*scale) @ k^T (k natural layout is already B^T) ---
  v16h aq = load_a_frag(q_h + ((size_t)(b * NPTS + q0)) * CDIM + h * 32, CDIM);
  for (int j = 0; j < 64; ++j) {
    v16h bk = load_bt_frag(k_h + (size_t)(b2 * NPTS + j * 16) * 256 + h * 32, 256);
    v8f s = {};
    s = wmma_f16(aq, bk, s);
#pragma unroll
    for (int r = 0; r < 8; ++r)
      s_lds[(mbase + r) * 1024 + j * 16 + nidx] = s[r];
  }
  __syncthreads();

  // --- softmax: 2 lanes per row, 512 cols each ---
  int row = lane >> 1;
  int col0 = (lane & 1) * 512;
  float mx = -3.4e38f;
  for (int i = 0; i < 512; ++i) mx = fmaxf(mx, s_lds[row * 1024 + col0 + i]);
  mx = fmaxf(mx, __shfl_xor(mx, 1, 32));
  float sum = 0.f;
  for (int i = 0; i < 512; ++i) {
    float e = __expf(s_lds[row * 1024 + col0 + i] - mx);
    sum += e;
    p_lds[row * 1024 + col0 + i] = (half_t)e;
  }
  sum += __shfl_xor(sum, 1, 32);
  if ((lane & 1) == 0) inv_lds[row] = 1.0f / sum;
  __syncthreads();

  // --- PV and P @ ce_head (vT / ceT are [head-dim][key] = B^T layout) ---
  v8f av0 = {}, av1 = {}, ac = {};
  const half_t* vtb = vT + ((size_t)(b2 * 8 + h) * 32) * NPTS;
  const half_t* ctb = ceT + ((size_t)(b * 8 + h) * 8) * NPTS;
  for (int kk = 0; kk < 32; ++kk) {
    v16h p = load_a_frag(p_lds + kk * 32, 1024);
    v16h b0 = load_bt_frag(vtb + kk * 32, NPTS);
    v16h b1 = load_bt_frag(vtb + 16 * NPTS + kk * 32, NPTS);
    v16h bc = load_bt_frag_pad(ctb + kk * 32, NPTS, 8);
    av0 = wmma_f16(p, b0, av0);
    av1 = wmma_f16(p, b1, av1);
    ac  = wmma_f16(p, bc, ac);
  }
#pragma unroll
  for (int r = 0; r < 8; ++r) {
    int m = mbase + r;
    size_t rowg = (size_t)b * NPTS + q0 + m;
    float inv = inv_lds[m];
    xattn_h[rowg * CDIM + h * 32 + nidx]      = (half_t)(av0[r] * inv);
    xattn_h[rowg * CDIM + h * 32 + 16 + nidx] = (half_t)(av1[r] * inv);
    if (nidx < 8) {
      mdiff_h[rowg * MOT + h * 8 + nidx] =
          (half_t)(ac[r] * inv - ce[rowg * MOT + h * 8 + nidx]);
    }
  }
}

// ---------------- host orchestration ----------------

extern "C" void kernel_launch(void* const* d_in, const int* in_sizes, int n_in,
                              void* d_out, int out_size, void* d_ws, size_t ws_size,
                              hipStream_t stream) {
  const float* x     = (const float*)d_in[0];
  const float* cor   = (const float*)d_in[1];
  const float* g1    = (const float*)d_in[2];
  const float* b1    = (const float*)d_in[3];
  const float* g2    = (const float*)d_in[4];
  const float* b2    = (const float*)d_in[5];
  const float* Wq    = (const float*)d_in[6];
  const float* Wkv   = (const float*)d_in[7];
  const float* Wcor  = (const float*)d_in[8];
  const float* Wproj = (const float*)d_in[9];
  const float* bproj = (const float*)d_in[10];
  const float* Wm    = (const float*)d_in[11];
  const float* bm    = (const float*)d_in[12];
  const float* W1    = (const float*)d_in[13];
  const float* bfc1  = (const float*)d_in[14];
  const float* dww   = (const float*)d_in[15];
  const float* dwb   = (const float*)d_in[16];
  const float* pa    = (const float*)d_in[17];
  const float* W2    = (const float*)d_in[18];
  const float* bfc2  = (const float*)d_in[19];

  char* p = (char*)d_ws;
  auto alloc = [&](size_t bytes) { void* r = (void*)p; p += (bytes + 255) & ~(size_t)255; return r; };

  const size_t R = R_TOT;
  float*  xn      = (float*)alloc(R * CDIM * 4);
  half_t* xn_h    = (half_t*)alloc(R * CDIM * 2);
  float*  ce      = (float*)alloc(R * MOT * 4 + 256);
  half_t* ceT     = (half_t*)alloc(R * MOT * 2 + 256);
  half_t* q_h     = (half_t*)alloc(R * CDIM * 2);
  half_t* k_h     = (half_t*)alloc(R * CDIM * 2);
  half_t* vT      = (half_t*)alloc(R * CDIM * 2);
  half_t* xattn_h = (half_t*)alloc(R * CDIM * 2);
  half_t* mdiff_h = (half_t*)alloc(R * MOT * 2 + 256);
  float*  xn2     = (float*)alloc(R * CDIM * 4);
  half_t* back_h  = (half_t*)alloc(R * CDIM * 2);
  half_t* h_h     = (half_t*)alloc(R * HID * 2);
  half_t* WqT     = (half_t*)alloc(65536 * 2);
  half_t* WkvT    = (half_t*)alloc(131072 * 2);
  half_t* WprojT  = (half_t*)alloc(65536 * 2);
  half_t* WmT     = (half_t*)alloc(4096 * 2);
  half_t* W1T     = (half_t*)alloc(262144 * 2);
  half_t* W2T     = (half_t*)alloc(262144 * 2);
  float*  mu1     = (float*)alloc(CDIM * 4);
  float*  rs1     = (float*)alloc(CDIM * 4);
  float*  mu2     = (float*)alloc(CDIM * 4);
  float*  rs2     = (float*)alloc(CDIM * 4);

  // 1) BN1 stats + normalize/transpose + coord embed
  bn_stats_bcn<<<256, 256, 0, stream>>>(x, mu1, rs1);
  stage_norm1<<<R, 256, 0, stream>>>(x, cor, g1, b1, mu1, rs1, Wcor, xn, xn_h, ce, ceT);

  // 2) weight conversion to transposed f16
  cvt_transpose_h<<<(65536 + 255) / 256, 256, 0, stream>>>(Wq, WqT, 256, 256);
  cvt_transpose_h<<<(131072 + 255) / 256, 256, 0, stream>>>(Wkv, WkvT, 256, 512);
  cvt_transpose_h<<<(65536 + 255) / 256, 256, 0, stream>>>(Wproj, WprojT, 256, 256);
  cvt_transpose_h<<<(4096 + 255) / 256, 256, 0, stream>>>(Wm, WmT, 64, 64);
  cvt_transpose_h<<<(262144 + 255) / 256, 256, 0, stream>>>(W1, W1T, 256, 1024);
  cvt_transpose_h<<<(262144 + 255) / 256, 256, 0, stream>>>(W2, W2T, 1024, 256);

  // 3) q / kv projections (WMMA); softmax scale folded into q
  gemm_wmma<EPI_QSCALE><<<dim3(R / 64, 4), 128, 0, stream>>>(
      xn_h, WqT, R, 256, 256, nullptr, q_h, nullptr, nullptr, nullptr, nullptr, nullptr);
  gemm_wmma<EPI_KV><<<dim3(R / 64, 8), 128, 0, stream>>>(
      xn_h, WkvT, R, 512, 256, nullptr, k_h, vT, nullptr, nullptr, nullptr, nullptr);

  // 4) fused cross-frame attention
  attn_fused<<<dim3(64, 8, 16), 32, 0, stream>>>(q_h, k_h, vT, ceT, ce, xattn_h, mdiff_h);

  // 5) output projection + residual -> xn2
  gemm_wmma<EPI_F32_BIAS_RES><<<dim3(R / 64, 4), 128, 0, stream>>>(
      xattn_h, WprojT, R, 256, 256, xn2, nullptr, nullptr, bproj, xn, nullptr, nullptr);

  // 6) BN2 + MLP
  bn_stats_rc<<<256, 256, 0, stream>>>(xn2, mu2, rs2);
  bn2_apply<<<R, 256, 0, stream>>>(xn2, mu2, rs2, g2, b2, back_h);
  gemm_wmma<EPI_FC1><<<dim3(R / 64, 16), 128, 0, stream>>>(
      back_h, W1T, R, 1024, 256, nullptr, h_h, nullptr, bfc1, dww, dwb, pa);
  gemm_wmma<EPI_FC2_TRANS><<<dim3(R / 64, 4), 128, 0, stream>>>(
      h_h, W2T, R, 256, 1024, (float*)d_out, nullptr, nullptr, bfc2, x, nullptr, nullptr);

  // 7) motion head: (crev - ce) @ Wm + bm
  gemm_wmma<EPI_F32_BIAS><<<dim3(R / 64, 1), 128, 0, stream>>>(
      mdiff_h, WmT, R, 64, 64, (float*)d_out + 16 * 256 * 1024, nullptr, nullptr, bm,
      nullptr, nullptr, nullptr);

  (void)in_sizes; (void)n_in; (void)out_size; (void)ws_size;
}
